// test_LSTM_74534862455048
// MI455X (gfx1250) — matrix operand order repaired and checked by reading, not compiled, not run on hardware
//
#include <hip/hip_runtime.h>
#include <math.h>

// ---------------------------------------------------------------------------
// Routed 2-cell LSTM: per-step GEMM gates = Wcat(bf16) @ [x;h](bf16) via
// v_wmma_f32_16x16x32_bf16 (f32 accumulate), then f32 pointwise LSTM update.
// B=64, S=256, E=H=1024, C=2, 4H=4096.  Gate rows: row = c*4096 + gate*1024 + h.
// GEMM v2: wave = 1 row-tile x 2 col-tiles (A reuse in registers);
//          block = 8 row-tiles x 32 tokens, activations staged in LDS.
// ---------------------------------------------------------------------------

#define Bz   64
#define Sz   256
#define Ez   1024
#define Hz   1024
#define ROWS 8192          // C * 4H
#define KDIM 2048          // E + H
#define LROW 2056          // LDS row pitch (pad 8 elems = 4-bank skew)

typedef __attribute__((ext_vector_type(16))) __bf16 v16bf;
typedef __attribute__((ext_vector_type(8)))  __bf16 v8bf;
typedef __attribute__((ext_vector_type(8)))  float  v8f;

// ---------- weight conversion: Wcat[row][k] bf16, row = c*4096+g, k = [E | H]
__global__ __launch_bounds__(256) void lstm_wconv(
    const float* __restrict__ W_ih, const float* __restrict__ W_hh,
    __bf16* __restrict__ Wcat)
{
    // 8192 rows * 2048 cols / 8 per thread = 2,097,152 threads
    size_t idx   = (size_t)blockIdx.x * 256 + threadIdx.x;
    int    chunk = (int)(idx & 255);          // 256 chunks of 8 per row
    size_t row   = idx >> 8;                  // 0..8191
    int    k     = chunk * 8;
    const float* src = (k < Ez)
        ? (W_ih + row * (size_t)Ez + k)
        : (W_hh + row * (size_t)Hz + (k - Ez));
    __bf16* dst = Wcat + row * (size_t)KDIM + k;
#pragma unroll
    for (int j = 0; j < 8; ++j) dst[j] = (__bf16)src[j];
}

// ---------- one-time prep: bias sum, cid routing table, c-state=0, act(t=0)
__global__ __launch_bounds__(256) void lstm_prep(
    const int* __restrict__ input, const int* __restrict__ cell_map,
    const float* __restrict__ b_ih, const float* __restrict__ b_hh,
    const float* __restrict__ embed,
    float* __restrict__ bias, int* __restrict__ cid,
    float* __restrict__ cT, __bf16* __restrict__ act)
{
    int idx = blockIdx.x * 256 + threadIdx.x;          // 0..131071
    if (idx < ROWS)      bias[idx] = b_ih[idx] + b_hh[idx];
    if (idx < Bz * Sz)   cid[idx]  = cell_map[input[idx]];
    if (idx < Bz * Hz)   cT[idx]   = 0.0f;
    // act init (B*KDIM = 131072 elements): x = embed[:,0,:], h = 0
    int b = idx >> 11, k = idx & (KDIM - 1);
    act[idx] = (k < Ez) ? (__bf16)embed[((size_t)b * Sz) * Ez + k] : (__bf16)0.0f;
}

// ---------- per-step GEMM: G[token][row] = sum_k Wcat[row][k] * act[token][k]
// grid: 128 blocks = (512 row-tiles / 8 per block) x 2 token-groups
__global__ __launch_bounds__(256) void lstm_gemm_gates(
    const __bf16* __restrict__ W, const __bf16* __restrict__ act,
    float* __restrict__ G)
{
    extern __shared__ __bf16 sact[];            // [32][LROW] staged activations
    const int tid  = threadIdx.x;
    const int lane = tid & 31;
    const int wave = tid >> 5;
    const int bid  = blockIdx.x;
    const int cgrp = bid & 1;                   // token group: 32*cgrp .. +31
    const int rt   = (bid >> 1) * 8 + wave;     // row-tile 0..511
    const int m0   = rt * 16;

    // ---- stage 32 tokens x 2048 K into LDS: 8192 chunks of 16 B
    {
        const __bf16* __restrict__ src = act + (size_t)(32 * cgrp) * KDIM;
#pragma unroll
        for (int i = 0; i < 32; ++i) {
            int ch     = i * 256 + tid;
            int tok    = ch >> 8;               // 0..31
            int within = ch & 255;              // 16B chunk within row
            *(v8bf*)(sact + tok * LROW + within * 8) =
                *(const v8bf*)(src + (size_t)tok * KDIM + within * 8);
        }
    }
    __syncthreads();

    // A fragment addressing (16x32 bf16, ISA §7.12.2)
    const int half  = lane >> 4;                // 0: lanes 0-15, 1: 16-31
    const int aKoff = half * 8;
    const __bf16* __restrict__ arow = W + (size_t)(m0 + (lane & 15)) * KDIM;
    // B fragment addressing (32x16 bf16): column = lane&15 (+16 for tile 1)
    const int bKoff = half * 8;
    const __bf16* __restrict__ bc0 = sact + (lane & 15) * LROW;
    const __bf16* __restrict__ bc1 = sact + ((lane & 15) + 16) * LROW;

    v8f acc0 = {}, acc1 = {};
#pragma unroll 4
    for (int kk = 0; kk < KDIM / 32; ++kk) {
        const int k0 = kk * 32;
        __builtin_prefetch(arow + k0 + 256, 0, 3);   // global_prefetch_b8
        v8bf alo = *(const v8bf*)(arow + k0 + aKoff);
        v8bf ahi = *(const v8bf*)(arow + k0 + 16 + aKoff);
        v16bf a = __builtin_shufflevector(alo, ahi,
            0, 1, 2, 3, 4, 5, 6, 7, 8, 9, 10, 11, 12, 13, 14, 15);
        // LDS B fragments as 2x16B (row pitch is 16B- but not 32B-aligned)
        v8bf b0l = *(const v8bf*)(bc0 + k0 + bKoff);
        v8bf b0h = *(const v8bf*)(bc0 + k0 + bKoff + 16);
        v8bf b1l = *(const v8bf*)(bc1 + k0 + bKoff);
        v8bf b1h = *(const v8bf*)(bc1 + k0 + bKoff + 16);
        v16bf b0 = __builtin_shufflevector(b0l, b0h,
            0, 1, 2, 3, 4, 5, 6, 7, 8, 9, 10, 11, 12, 13, 14, 15);
        v16bf b1 = __builtin_shufflevector(b1l, b1h,
            0, 1, 2, 3, 4, 5, 6, 7, 8, 9, 10, 11, 12, 13, 14, 15);
        acc0 = __builtin_amdgcn_wmma_f32_16x16x32_bf16(
            false, a, false, b0, (short)0, acc0, false, false);
        acc1 = __builtin_amdgcn_wmma_f32_16x16x32_bf16(
            false, a, false, b1, (short)0, acc1, false, false);
    }
    // D: lane holds rows m0+half*8 .. +7 (contiguous) of its column
    float* g0 = G + (size_t)(32 * cgrp + (lane & 15)) * ROWS + m0 + half * 8;
    *(v8f*)g0 = acc0;
    *(v8f*)(g0 + (size_t)16 * ROWS) = acc1;
}

// ---------- per-step pointwise LSTM + state/act update
__global__ __launch_bounds__(256) void lstm_pointwise(
    const float* __restrict__ G, const float* __restrict__ bias,
    const int* __restrict__ cid, const float* __restrict__ embed,
    float* __restrict__ outs, float* __restrict__ hT, float* __restrict__ cT,
    __bf16* __restrict__ act, int t)
{
    int idx = blockIdx.x * 256 + threadIdx.x;   // 0..65535
    int b = idx >> 10;
    int h = idx & (Hz - 1);
    int c = cid[b * Sz + t];

    size_t gb = (size_t)b * ROWS + (size_t)c * 4096 + h;
    size_t bb = (size_t)c * 4096 + h;
    float gi = G[gb       ] + bias[bb       ];
    float gf = G[gb + 1*Hz] + bias[bb + 1*Hz];
    float gg = G[gb + 2*Hz] + bias[bb + 2*Hz];
    float go = G[gb + 3*Hz] + bias[bb + 3*Hz];

    float iv = 1.0f / (1.0f + expf(-gi));
    float fv = 1.0f / (1.0f + expf(-gf));
    float ov = 1.0f / (1.0f + expf(-go));

    float cold = cT[idx];
    float cnew = fv * cold + iv * tanhf(gg);
    float hnew = ov * tanhf(cnew);

    outs[((size_t)b * Sz + t) * Hz + h] = hnew;
    cT[idx] = cnew;
    if (t == Sz - 1) hT[idx] = hnew;

    // pack activations for next step: act[b][0..1023]=x_{t+1}, [1024..2047]=h
    act[(size_t)b * KDIM + Ez + h] = (__bf16)hnew;
    if (t + 1 < Sz)
        act[(size_t)b * KDIM + h] =
            (__bf16)embed[((size_t)b * Sz + (t + 1)) * Ez + h];
}

// ---------------------------------------------------------------------------
extern "C" void kernel_launch(void* const* d_in, const int* in_sizes, int n_in,
                              void* d_out, int out_size, void* d_ws, size_t ws_size,
                              hipStream_t stream)
{
    const int*   input  = (const int*)  d_in[0];   // [B,S]
    const float* embed  = (const float*)d_in[1];   // [B,S,E]
    const int*   cmap   = (const int*)  d_in[2];   // [V]
    const float* W_ih   = (const float*)d_in[3];   // [C,4H,E]
    const float* W_hh   = (const float*)d_in[4];   // [C,4H,H]
    const float* b_ih   = (const float*)d_in[5];   // [C,4H]
    const float* b_hh   = (const float*)d_in[6];   // [C,4H]

    float* outs = (float*)d_out;                       // [B,S,H]
    float* hT   = outs + (size_t)Bz * Sz * Hz;         // [B,H] final h
    float* cT   = hT + (size_t)Bz * Hz;                // [B,H] final c (= state)

    // workspace layout (all offsets 256B aligned)
    char* ws = (char*)d_ws;
    __bf16* Wcat = (__bf16*)(ws);                              // 33,554,432 B
    __bf16* act  = (__bf16*)(ws + 33554432);                   //    262,144 B
    float*  G    = (float*) (ws + 33554432 + 262144);          //  2,097,152 B
    float*  bias = (float*) (ws + 33554432 + 262144 + 2097152);//     32,768 B
    int*    cid  = (int*)   (ws + 33554432 + 262144 + 2097152 + 32768); // 65,536 B

    const size_t lds_bytes = (size_t)32 * LROW * sizeof(__bf16);  // 131,584 B

    lstm_wconv<<<8192, 256, 0, stream>>>(W_ih, W_hh, Wcat);
    lstm_prep <<<512,  256, 0, stream>>>(input, cmap, b_ih, b_hh, embed,
                                         bias, cid, cT, act);
    for (int t = 0; t < Sz; ++t) {
        lstm_gemm_gates<<<128, 256, lds_bytes, stream>>>(Wcat, act, G);
        lstm_pointwise <<<256, 256, 0, stream>>>(G, bias, cid, embed,
                                                 outs, hT, cT, act, t);
    }
    (void)in_sizes; (void)n_in; (void)out_size; (void)ws_size;
}
